// DeltaNet_30743375905046
// MI455X (gfx1250) — matrix operand order repaired and checked by reading, not compiled, not run on hardware
//
#include <hip/hip_runtime.h>
#include <cstddef>
#include <cstdint>

// ---------------- problem dims ----------------
constexpr int Bn   = 4;
constexpr int Ln   = 4096;
constexpr int Dn   = 1024;
constexpr int Hn   = 4;
constexpr int DHn  = 256;   // head dim
constexpr int Cn   = 64;    // chunk size
constexpr int NCn  = Ln / Cn;
constexpr int Mtok = Bn * Ln;       // 16384 tokens

typedef __attribute__((ext_vector_type(16))) _Float16 v16h;
typedef __attribute__((ext_vector_type(8)))  _Float16 v8h;
typedef __attribute__((ext_vector_type(8)))  float    v8f;

// ---------------- WMMA fragment helpers (wave32, 16x16x32 f16) ----------------
// A 16x32 f16 (ISA 7.12.2): lane m = lane&15; lanes 0-15 hold K {0..7,16..23},
// lanes 16-31 hold K {8..15,24..31}. Storage row-major [m][k], k contiguous.
__device__ static inline v16h ld_a(const _Float16* p, int tm, int k0, int ld, int lane) {
  int m  = tm + (lane & 15);
  int kb = k0 + ((lane >> 4) << 3);
  const _Float16* q = p + (size_t)m * ld + kb;
  union { v8h h[2]; v16h v; } r;
  r.h[0] = *(const v8h*)(q);
  r.h[1] = *(const v8h*)(q + 16);
  return r.v;
}
// A operand where storage is k-major: element (m,k) lives at p[k*ld + m] (for K^T uses).
__device__ static inline v16h ld_a_kmaj(const _Float16* p, int tm, int k0, int ld, int lane) {
  int m  = tm + (lane & 15);
  int kb = k0 + ((lane >> 4) << 3);
  union { _Float16 h[16]; v16h v; } r;
#pragma unroll
  for (int e = 0; e < 16; ++e) {
    int k = kb + (e & 7) + ((e >> 3) << 4);
    r.h[e] = p[(size_t)k * ld + m];
  }
  return r.v;
}
// B 32x16 f16: lane n = lane&15; lanes 0-15 hold K {0..7,16..23}, lanes 16-31 hold K {8..15,24..31}
// (same as A). Storage n-major [n][k], k contiguous.
__device__ static inline v16h ld_b(const _Float16* p, int tn, int k0, int ld, int lane) {
  int n  = tn + (lane & 15);
  int kb = k0 + ((lane >> 4) << 3);
  const _Float16* q = p + (size_t)n * ld + kb;
  union { v8h h[2]; v16h v; } r;
  r.h[0] = *(const v8h*)(q);
  r.h[1] = *(const v8h*)(q + 16);
  return r.v;
}
__device__ static inline v8f wmma16(v16h a, v16h b, v8f c) {
  return __builtin_amdgcn_wmma_f32_16x16x32_f16(false, a, false, b, (short)0, c, false, false);
}

// CDNA5 async global->LDS DMA (ASYNCcnt-tracked, GVS mode: SGPR base + 32b VGPR offset).
__device__ static inline void async_g2l_b128(uint32_t lds_byte, const void* gbase, uint32_t goff) {
  asm volatile("global_load_async_to_lds_b128 %0, %1, %2"
               :: "v"(lds_byte), "v"(goff), "s"(gbase) : "memory");
}
__device__ static inline void wait_async0() {
  asm volatile("s_wait_asynccnt 0" ::: "memory");
}

// ---------------- small elementwise kernels ----------------
__global__ __launch_bounds__(256) void k_cvt(const float* __restrict__ x, _Float16* __restrict__ y) {
  size_t i = (size_t)blockIdx.x * 256 + threadIdx.x;
  y[i] = (_Float16)x[i];
}
// Wt[n][k] = f16(W[k][n]); W is [D,D] row-major (in,out)
__global__ __launch_bounds__(256) void k_wT(const float* __restrict__ W, _Float16* __restrict__ Wt) {
  size_t i = (size_t)blockIdx.x * 256 + threadIdx.x;
  int n = (int)(i >> 10), k = (int)(i & 1023);
  Wt[i] = (_Float16)W[(size_t)k * Dn + n];
}

// ---------------- WMMA GEMM: C[M,N] = A[M,K] * Bt[N,K]^T ----------------
template <typename OUT_T>
__global__ __launch_bounds__(256) void k_gemm(const _Float16* __restrict__ Ag,
                                              const _Float16* __restrict__ Bg,
                                              OUT_T* __restrict__ Cg,
                                              int Md, int Nd, int Kd) {
  __shared__ _Float16 As[128][72];
  __shared__ _Float16 Bs[128][72];
  const int tid  = threadIdx.x;
  const int lane = tid & 31;
  const int wid  = tid >> 5;
  const int wm = wid >> 1, wn = wid & 1;
  const int bm = blockIdx.y * 128;
  const int bn = blockIdx.x * 128;
  (void)Md;

  v8f acc[2][4];
#pragma unroll
  for (int i = 0; i < 2; ++i)
#pragma unroll
    for (int j = 0; j < 4; ++j)
      acc[i][j] = v8f{0.f,0.f,0.f,0.f,0.f,0.f,0.f,0.f};

  const int nk = Kd >> 6;
  for (int kt = 0; kt < nk; ++kt) {
#pragma unroll
    for (int i = 0; i < 4; ++i) {
      int idx = tid + (i << 8);          // 0..1023
      int r = idx >> 3, c8 = idx & 7;
      *(v8h*)&As[r][c8 << 3] = *(const v8h*)(Ag + (size_t)(bm + r) * Kd + (kt << 6) + (c8 << 3));
      *(v8h*)&Bs[r][c8 << 3] = *(const v8h*)(Bg + (size_t)(bn + r) * Kd + (kt << 6) + (c8 << 3));
    }
    if (kt + 1 < nk) {  // prefetch next K tile into cache
      __builtin_prefetch(Ag + (size_t)(bm + (tid >> 1)) * Kd + ((kt + 1) << 6) + ((tid & 1) << 5), 0, 0);
      __builtin_prefetch(Bg + (size_t)(bn + (tid >> 1)) * Kd + ((kt + 1) << 6) + ((tid & 1) << 5), 0, 0);
    }
    __syncthreads();
#pragma unroll
    for (int kk = 0; kk < 64; kk += 32) {
      v16h a0 = ld_a(&As[0][0], wm * 32,      kk, 72, lane);
      v16h a1 = ld_a(&As[0][0], wm * 32 + 16, kk, 72, lane);
      v16h b0 = ld_b(&Bs[0][0], wn * 64,      kk, 72, lane);
      v16h b1 = ld_b(&Bs[0][0], wn * 64 + 16, kk, 72, lane);
      v16h b2 = ld_b(&Bs[0][0], wn * 64 + 32, kk, 72, lane);
      v16h b3 = ld_b(&Bs[0][0], wn * 64 + 48, kk, 72, lane);
      acc[0][0] = wmma16(a0, b0, acc[0][0]); acc[0][1] = wmma16(a0, b1, acc[0][1]);
      acc[0][2] = wmma16(a0, b2, acc[0][2]); acc[0][3] = wmma16(a0, b3, acc[0][3]);
      acc[1][0] = wmma16(a1, b0, acc[1][0]); acc[1][1] = wmma16(a1, b1, acc[1][1]);
      acc[1][2] = wmma16(a1, b2, acc[1][2]); acc[1][3] = wmma16(a1, b3, acc[1][3]);
    }
    __syncthreads();
  }
#pragma unroll
  for (int i = 0; i < 2; ++i)
#pragma unroll
    for (int j = 0; j < 4; ++j) {
      int tm = bm + wm * 32 + i * 16;
      int tn = bn + wn * 64 + j * 16;
      int n  = tn + (lane & 15);
      int m0 = tm + ((lane >> 4) << 3);
#pragma unroll
      for (int r = 0; r < 8; ++r)
        Cg[(size_t)(m0 + r) * Nd + n] = (OUT_T)acc[i][j][r];
    }
}

// ---------------- causal depthwise conv (K=4) + SiLU (+ per-head L2 norm) ----------------
// in: lin [B,L,D] f16; out: [B,H,L,Dh] f16
__global__ __launch_bounds__(256) void k_conv(const _Float16* __restrict__ lin,
                                              const float* __restrict__ w,
                                              _Float16* __restrict__ out, int do_norm) {
  __shared__ float red[256];
  int tok = blockIdx.x;
  int b = tok >> 12, l = tok & 4095;
  int t = threadIdx.x;
  float y[4];
#pragma unroll
  for (int h = 0; h < Hn; ++h) {
    int ch = h * DHn + t;
    float acc = 0.f;
#pragma unroll
    for (int j = 0; j < 4; ++j) {
      int ll = l + j - 3;
      if (ll >= 0) acc += w[ch * 4 + j] * (float)lin[((size_t)b * Ln + ll) * Dn + ch];
    }
    y[h] = acc / (1.f + __expf(-acc));   // SiLU
  }
  if (do_norm) {
#pragma unroll
    for (int h = 0; h < Hn; ++h) {
      red[t] = y[h] * y[h];
      __syncthreads();
      for (int s = 128; s > 0; s >>= 1) { if (t < s) red[t] += red[t + s]; __syncthreads(); }
      float nrm = sqrtf(red[0]);
      __syncthreads();
      y[h] *= 1.f / fmaxf(nrm, 1e-12f);
    }
  }
#pragma unroll
  for (int h = 0; h < Hn; ++h)
    out[(((size_t)(b * Hn + h) * Ln) + l) * DHn + t] = (_Float16)y[h];
}

// ---------------- beta = sigmoid(x @ Wb), stored [B,H,L] f32 ----------------
__global__ __launch_bounds__(256) void k_beta(const float* __restrict__ x,
                                              const float* __restrict__ Wb,
                                              float* __restrict__ beta) {
  __shared__ float red[256];
  int tok = blockIdx.x;
  int b = tok >> 12, l = tok & 4095;
  int t = threadIdx.x;
  float acc[4] = {0.f, 0.f, 0.f, 0.f};
  const float* xr = x + (size_t)tok * Dn;
  for (int d = t; d < Dn; d += 256) {
    float xv = xr[d];
#pragma unroll
    for (int h = 0; h < Hn; ++h) acc[h] += xv * Wb[d * Hn + h];
  }
#pragma unroll
  for (int h = 0; h < Hn; ++h) {
    red[t] = acc[h];
    __syncthreads();
    for (int s = 128; s > 0; s >>= 1) { if (t < s) red[t] += red[t + s]; __syncthreads(); }
    if (t == 0) beta[((size_t)(b * Hn + h)) * Ln + l] = 1.f / (1.f + __expf(-red[0]));
    __syncthreads();
  }
}

// ---------------- delta-rule chunk scan: one block per (b,h); S resident in LDS ----------------
constexpr size_t SM_S  = 0;        // f16 [256][256], Sb[d2][d1] = S[d1][d2]
constexpr size_t SM_KC = 131072;   // f16 [64][256]
constexpr size_t SM_UB = 163840;   // f16 [64][256] (k*beta) ; later Ub f16 [256][64]
constexpr size_t SM_UF = 196608;   // f32 [256][64]  U stored [e][c]
constexpr size_t SM_A  = 262144;   // f32 [64][64] A ; later f16 [64][64] attn
constexpr size_t SM_QC = 278528;   // f16 [64][256]
constexpr size_t SM_BI = 311296;   // f32 [64]
constexpr size_t SMEM_DELTA = 311552;  // 304.25 KB dynamic LDS (<= 320KB/WGP)

__global__ __launch_bounds__(256) void k_delta(const _Float16* __restrict__ qh,
                                               const _Float16* __restrict__ kh,
                                               const _Float16* __restrict__ vh,
                                               const float* __restrict__ beta,
                                               float* __restrict__ og) {
  extern __shared__ char smem[];
  _Float16* Sb   = (_Float16*)(smem + SM_S);
  _Float16* kc   = (_Float16*)(smem + SM_KC);
  _Float16* kbc  = (_Float16*)(smem + SM_UB);
  _Float16* Ub   = (_Float16*)(smem + SM_UB);
  float*    Uf   = (float*)   (smem + SM_UF);
  float*    Am   = (float*)   (smem + SM_A);
  _Float16* attn = (_Float16*)(smem + SM_A);
  _Float16* qc   = (_Float16*)(smem + SM_QC);
  float*    bi   = (float*)   (smem + SM_BI);

  const int tid  = threadIdx.x;
  const int lane = tid & 31;
  const int wid  = tid >> 5;
  const int bh   = blockIdx.x;
  const _Float16* qg = qh + (size_t)bh * Ln * DHn;
  const _Float16* kg = kh + (size_t)bh * Ln * DHn;
  const _Float16* vg = vh + (size_t)bh * Ln * DHn;
  const float*    bg = beta + (size_t)bh * Ln;
  float*          ogb = og + (size_t)bh * Ln * DHn;

  for (int i = tid; i < DHn * DHn; i += 256) Sb[i] = (_Float16)0.f;
  __syncthreads();

  for (int nc = 0; nc < NCn; ++nc) {
    const size_t base = (size_t)nc * Cn * DHn;
    // P1: async-DMA the k,q chunk tiles into LDS (16B/lane, ASYNCcnt tracked),
    // overlap with the beta-row load; then s_wait_asynccnt + barrier.
    {
      const uint32_t gb = (uint32_t)(base * sizeof(_Float16));
#pragma unroll
      for (int it = 0; it < (Cn * DHn) / (8 * 256); ++it) {   // 8 iters
        int i = tid + (it << 8);
        uint32_t go = gb + (uint32_t)i * 16u;
        async_g2l_b128((uint32_t)SM_KC + (uint32_t)i * 16u, (const void*)kg, go);
        async_g2l_b128((uint32_t)SM_QC + (uint32_t)i * 16u, (const void*)qg, go);
      }
    }
    if (tid < Cn) bi[tid] = bg[nc * Cn + tid];
    wait_async0();
    __syncthreads();
    for (int i = tid; i < Cn * DHn; i += 256)
      kbc[i] = (_Float16)((float)kc[i] * bi[i >> 8]);
    __syncthreads();

    // P2: A = (kb @ k^T) .* strict_lower  [64x64]
    for (int it = 0; it < 2; ++it) {
      int t = wid * 2 + it;
      int tm = (t >> 2) << 4, tn = (t & 3) << 4;
      v8f acc = {0.f,0.f,0.f,0.f,0.f,0.f,0.f,0.f};
      for (int k0 = 0; k0 < DHn; k0 += 32)
        acc = wmma16(ld_a(kbc, tm, k0, DHn, lane), ld_b(kc, tn, k0, DHn, lane), acc);
      int n = tn + (lane & 15);
      int m0 = tm + ((lane >> 4) << 3);
#pragma unroll
      for (int r = 0; r < 8; ++r) { int m = m0 + r; Am[m * 64 + n] = (m > n) ? acc[r] : 0.f; }
    }
    __syncthreads();

    // P3: rhs = v*beta - kb @ S   -> Uf[e][c]
    for (int it = 0; it < 8; ++it) {
      int t = wid * 8 + it;
      int tm = (t >> 4) << 4, tn = (t & 15) << 4;
      v8f acc = {0.f,0.f,0.f,0.f,0.f,0.f,0.f,0.f};
      for (int k0 = 0; k0 < DHn; k0 += 32)
        acc = wmma16(ld_a(kbc, tm, k0, DHn, lane), ld_b(Sb, tn, k0, DHn, lane), acc);
      int e = tn + (lane & 15);
      int m0 = tm + ((lane >> 4) << 3);
#pragma unroll
      for (int r = 0; r < 8; ++r) {
        int c = m0 + r;
        float vb = (float)vg[base + (size_t)c * DHn + e] * bi[c];
        Uf[(size_t)e * 64 + c] = vb - acc[r];
      }
    }
    __syncthreads();

    // P4: forward substitution (I+A) U = rhs ; one RHS column per lane, no sync needed
    {
      float* Ue = Uf + (size_t)tid * 64;
      for (int t2 = 1; t2 < Cn; ++t2) {
        float a = Ue[t2];
        const float* Ar = Am + t2 * 64;
        for (int s = 0; s < t2; ++s) a -= Ar[s] * Ue[s];
        Ue[t2] = a;
      }
    }
    __syncthreads();

    // P5: Ub = f16(U); attn = (q @ k^T) .* incl_lower (reuses A region)
    for (int i = tid; i < Cn * DHn; i += 256) Ub[i] = (_Float16)Uf[i];
    for (int it = 0; it < 2; ++it) {
      int t = wid * 2 + it;
      int tm = (t >> 2) << 4, tn = (t & 3) << 4;
      v8f acc = {0.f,0.f,0.f,0.f,0.f,0.f,0.f,0.f};
      for (int k0 = 0; k0 < DHn; k0 += 32)
        acc = wmma16(ld_a(qc, tm, k0, DHn, lane), ld_b(kc, tn, k0, DHn, lane), acc);
      int n = tn + (lane & 15);
      int m0 = tm + ((lane >> 4) << 3);
#pragma unroll
      for (int r = 0; r < 8; ++r) {
        int m = m0 + r;
        attn[m * 64 + n] = (m >= n) ? (_Float16)acc[r] : (_Float16)0.f;
      }
    }
    __syncthreads();

    // P6: o = q @ S + attn @ U  -> global [B,H,L,Dh] f32
    for (int it = 0; it < 8; ++it) {
      int t = wid * 8 + it;
      int tm = (t >> 4) << 4, tn = (t & 15) << 4;
      v8f acc = {0.f,0.f,0.f,0.f,0.f,0.f,0.f,0.f};
      for (int k0 = 0; k0 < DHn; k0 += 32)
        acc = wmma16(ld_a(qc, tm, k0, DHn, lane), ld_b(Sb, tn, k0, DHn, lane), acc);
      for (int k0 = 0; k0 < Cn; k0 += 32)
        acc = wmma16(ld_a(attn, tm, k0, Cn, lane), ld_b(Ub, tn, k0, Cn, lane), acc);
      int e = tn + (lane & 15);
      int m0 = tm + ((lane >> 4) << 3);
#pragma unroll
      for (int r = 0; r < 8; ++r)
        ogb[base + (size_t)(m0 + r) * DHn + e] = acc[r];
    }
    __syncthreads();

    // P7: S += k^T @ U  (C-operand tiles read/modify/write Sb with b128 vectors)
    for (int it = 0; it < 32; ++it) {
      int t = wid * 32 + it;
      int tm = (t >> 4) << 4, tn = (t & 15) << 4;
      int n = tn + (lane & 15);
      int m0 = tm + ((lane >> 4) << 3);
      union { v8h v; _Float16 h[8]; } cs;
      cs.v = *(const v8h*)&Sb[(size_t)n * DHn + m0];
      v8f acc;
#pragma unroll
      for (int r = 0; r < 8; ++r) acc[r] = (float)cs.h[r];
      for (int k0 = 0; k0 < Cn; k0 += 32)
        acc = wmma16(ld_a_kmaj(kc, tm, k0, DHn, lane), ld_b(Ub, tn, k0, Cn, lane), acc);
#pragma unroll
      for (int r = 0; r < 8; ++r) cs.h[r] = (_Float16)acc[r];
      *(v8h*)&Sb[(size_t)n * DHn + m0] = cs.v;
    }
    __syncthreads();
  }
}

// ---------------- RMSNorm + swish gate -> f16 [B,L,D] ----------------
__global__ __launch_bounds__(256) void k_gate(const float* __restrict__ od,
                                              const _Float16* __restrict__ glin,
                                              const float* __restrict__ norm_w,
                                              _Float16* __restrict__ ob) {
  __shared__ float red[256];
  int tok = blockIdx.x;
  int b = tok >> 12, l = tok & 4095;
  int t = threadIdx.x;
#pragma unroll
  for (int h = 0; h < Hn; ++h) {
    float o = od[(((size_t)(b * Hn + h)) * Ln + l) * DHn + t];
    red[t] = o * o;
    __syncthreads();
    for (int s = 128; s > 0; s >>= 1) { if (t < s) red[t] += red[t + s]; __syncthreads(); }
    float var = red[0] * (1.f / 256.f);
    __syncthreads();
    float rr = rsqrtf(var + 1e-5f);
    float g = (float)glin[(size_t)tok * Dn + h * DHn + t];
    float val = o * rr * norm_w[t] * g / (1.f + __expf(-g));
    ob[(size_t)tok * Dn + h * DHn + t] = (_Float16)val;
  }
}

// ---------------- workspace arena ----------------
constexpr size_t SZ_TOKD_H = (size_t)Mtok * Dn * 2;  // 32 MB (f16 [16384,1024])
constexpr size_t SZ_W_H    = (size_t)Dn * Dn * 2;    // 2 MB
constexpr size_t OFF_XB   = 0;
constexpr size_t OFF_WT   = OFF_XB + SZ_TOKD_H;
constexpr size_t OFF_LIN0 = OFF_WT + 5 * SZ_W_H;     // q_lin, then g_lin
constexpr size_t OFF_LIN1 = OFF_LIN0 + SZ_TOKD_H;    // k_lin; later o_delta f32 (spans LIN1+LIN2)
constexpr size_t OFF_LIN2 = OFF_LIN1 + SZ_TOKD_H;    // v_lin
constexpr size_t OFF_QH   = OFF_LIN2 + SZ_TOKD_H;    // qh; later gated-out f16
constexpr size_t OFF_KH   = OFF_QH + SZ_TOKD_H;
constexpr size_t OFF_VH   = OFF_KH + SZ_TOKD_H;
constexpr size_t OFF_BETA = OFF_VH + SZ_TOKD_H;      // f32 [B,H,L]

extern "C" void kernel_launch(void* const* d_in, const int* in_sizes, int n_in,
                              void* d_out, int out_size, void* d_ws, size_t ws_size,
                              hipStream_t stream) {
  (void)in_sizes; (void)n_in; (void)out_size; (void)ws_size;
  const float* x  = (const float*)d_in[0];
  const float* Wq = (const float*)d_in[1];
  const float* Wk = (const float*)d_in[2];
  const float* Wv = (const float*)d_in[3];
  const float* Wb = (const float*)d_in[4];
  const float* Wg = (const float*)d_in[5];
  const float* Wo = (const float*)d_in[6];
  const float* cq = (const float*)d_in[7];
  const float* ck = (const float*)d_in[8];
  const float* cv = (const float*)d_in[9];
  const float* nw = (const float*)d_in[10];

  char* ws = (char*)d_ws;
  _Float16* xb   = (_Float16*)(ws + OFF_XB);
  _Float16* Wtq  = (_Float16*)(ws + OFF_WT + 0 * SZ_W_H);
  _Float16* Wtk  = (_Float16*)(ws + OFF_WT + 1 * SZ_W_H);
  _Float16* Wtv  = (_Float16*)(ws + OFF_WT + 2 * SZ_W_H);
  _Float16* Wtg  = (_Float16*)(ws + OFF_WT + 3 * SZ_W_H);
  _Float16* Wto  = (_Float16*)(ws + OFF_WT + 4 * SZ_W_H);
  _Float16* lin0 = (_Float16*)(ws + OFF_LIN0);
  _Float16* lin1 = (_Float16*)(ws + OFF_LIN1);
  _Float16* lin2 = (_Float16*)(ws + OFF_LIN2);
  _Float16* qhp  = (_Float16*)(ws + OFF_QH);
  _Float16* khp  = (_Float16*)(ws + OFF_KH);
  _Float16* vhp  = (_Float16*)(ws + OFF_VH);
  float*    betap = (float*)(ws + OFF_BETA);
  float*    odp   = (float*)(ws + OFF_LIN1);     // f32 [B,H,L,Dh], 64MB over LIN1+LIN2
  _Float16* obp   = (_Float16*)(ws + OFF_QH);    // gated output, reuses qh

  const int nwT = (Dn * Dn) / 256;
  k_wT<<<nwT, 256, 0, stream>>>(Wq, Wtq);
  k_wT<<<nwT, 256, 0, stream>>>(Wk, Wtk);
  k_wT<<<nwT, 256, 0, stream>>>(Wv, Wtv);
  k_wT<<<nwT, 256, 0, stream>>>(Wg, Wtg);
  k_wT<<<nwT, 256, 0, stream>>>(Wo, Wto);
  k_cvt<<<(Mtok * Dn) / 256, 256, 0, stream>>>(x, xb);

  dim3 gg(Dn / 128, Mtok / 128);
  k_gemm<_Float16><<<gg, 256, 0, stream>>>(xb, Wtq, lin0, Mtok, Dn, Dn);
  k_conv<<<Mtok, 256, 0, stream>>>(lin0, cq, qhp, 1);
  k_gemm<_Float16><<<gg, 256, 0, stream>>>(xb, Wtk, lin1, Mtok, Dn, Dn);
  k_conv<<<Mtok, 256, 0, stream>>>(lin1, ck, khp, 1);
  k_gemm<_Float16><<<gg, 256, 0, stream>>>(xb, Wtv, lin2, Mtok, Dn, Dn);
  k_conv<<<Mtok, 256, 0, stream>>>(lin2, cv, vhp, 0);
  k_beta<<<Mtok, 256, 0, stream>>>(x, Wb, betap);
  k_gemm<_Float16><<<gg, 256, 0, stream>>>(xb, Wtg, lin0, Mtok, Dn, Dn);  // g_lin

  hipFuncSetAttribute(reinterpret_cast<const void*>(k_delta),
                      hipFuncAttributeMaxDynamicSharedMemorySize, (int)SMEM_DELTA);
  k_delta<<<Bn * Hn, 256, SMEM_DELTA, stream>>>(qhp, khp, vhp, betap, odp);

  k_gate<<<Mtok, 256, 0, stream>>>(odp, lin0, nw, obp);
  k_gemm<float><<<gg, 256, 0, stream>>>(obp, Wto, (float*)d_out, Mtok, Dn, Dn);
}
